// GraphAttentionLayer_55018531061840
// MI455X (gfx1250) — matrix operand order repaired and checked by reading, not compiled, not run on hardware
//
#include <hip/hip_runtime.h>
#include <hip/hip_bf16.h>
#include <cstdint>
#include <cstddef>

// ---------------------------------------------------------------------------
// GAT layer for MI455X (gfx1250, wave32, WMMA).
//   Wh = h @ W                        (k1, f16 WMMA, f32 accum)
//   f1 = Wh a1, f2 = Wh a2, WhT f16   (k2)
//   Mb = LeakyReLU(max f1 + max f2)   (k3)  -- per-batch softmax shift bound
//   out = elu(softmax(mask(LR(f1+f2))) @ Wh)  (k4, fused single adj pass)
// adj (134 MB) is streamed exactly once -> HBM-bound ~6us @ 23.3 TB/s.
// ---------------------------------------------------------------------------

typedef __attribute__((ext_vector_type(16))) _Float16 v16h;
typedef __attribute__((ext_vector_type(8)))  _Float16 v8h;
typedef __attribute__((ext_vector_type(8)))  float    v8f;

#define GAT_ALPHA 0.2f
#define GAT_NEGINF (-9.0e15f)
#define Bn 8
#define Nn 2048
#define Fn 128

static __device__ __forceinline__ v16h load16h(const _Float16* p) {
  union { v16h v; struct { v8h lo; v8h hi; } s; } u;
  u.s.lo = *(const v8h*)p;
  u.s.hi = *(const v8h*)(p + 8);
  return u.v;
}

// ---------------- k1: Wh = h @ W (M=16384, N=128, K=128) -------------------
__global__ __launch_bounds__(128) void k1_wh(const float* __restrict__ h,
                                             const float* __restrict__ W,
                                             float* __restrict__ Wh) {
  __shared__ _Float16 WT[128 * 136];   // W transposed, f16, padded stride
  const int tid = threadIdx.x;
  for (int idx = tid; idx < 128 * 128; idx += 128) {
    int k = idx >> 7, n = idx & 127;
    WT[n * 136 + k] = (_Float16)W[idx];
  }
  __syncthreads();

  const int blk  = blockIdx.x;
  const int b    = blk >> 5;            // 32 tiles of 64 rows per batch
  const int tile = blk & 31;
  const int wave = tid >> 5;
  const int L    = tid & 31;
  const int m    = L & 15;
  const int half = L >> 4;
  const int i0   = tile * 64 + wave * 16;
  const int row  = i0 + m;

  const float* hp = h + ((size_t)(b * Nn + row)) * Fn;

  v8f zero8 = {0.f, 0.f, 0.f, 0.f, 0.f, 0.f, 0.f, 0.f};
  v8f acc[8];
#pragma unroll
  for (int t = 0; t < 8; ++t) acc[t] = zero8;

#pragma unroll
  for (int kk = 0; kk < 128; kk += 32) {
    const int kb = kk + half * 16;      // A layout: lane halves own K 0-15 / 16-31
    float4 ha = *(const float4*)(hp + kb);
    float4 hb = *(const float4*)(hp + kb + 4);
    float4 hc = *(const float4*)(hp + kb + 8);
    float4 hd = *(const float4*)(hp + kb + 12);
    v16h af;
    af[0]  = (_Float16)ha.x;  af[1]  = (_Float16)ha.y;
    af[2]  = (_Float16)ha.z;  af[3]  = (_Float16)ha.w;
    af[4]  = (_Float16)hb.x;  af[5]  = (_Float16)hb.y;
    af[6]  = (_Float16)hb.z;  af[7]  = (_Float16)hb.w;
    af[8]  = (_Float16)hc.x;  af[9]  = (_Float16)hc.y;
    af[10] = (_Float16)hc.z;  af[11] = (_Float16)hc.w;
    af[12] = (_Float16)hd.x;  af[13] = (_Float16)hd.y;
    af[14] = (_Float16)hd.z;  af[15] = (_Float16)hd.w;

    const int koff = kk + half * 16;    // B layout: lane halves own K 0-15 / 16-31
#pragma unroll
    for (int t = 0; t < 8; ++t) {
      const int n = t * 16 + m;
      v16h bf = load16h(&WT[n * 136 + koff]);
      acc[t] = __builtin_amdgcn_wmma_f32_16x16x32_f16(
          false, af, false, bf, (short)0, acc[t], false, false);
    }
  }

  float* whp = Wh + ((size_t)(b * Nn + i0)) * Fn;
#pragma unroll
  for (int t = 0; t < 8; ++t) {
#pragma unroll
    for (int r = 0; r < 8; ++r) {
      const int M = r + half * 8;       // C layout: VGPR r -> row r / r+8
      whp[(size_t)M * Fn + t * 16 + m] = acc[t][r];
    }
  }
}

// ---------------- k2: f1, f2, WhT(f16) per 32-row tile ---------------------
__global__ __launch_bounds__(256) void k2_prep(const float* __restrict__ Wh,
                                               const float* __restrict__ a,
                                               float* __restrict__ f1,
                                               float* __restrict__ f2,
                                               _Float16* __restrict__ WhT) {
  __shared__ float S[32 * 129];
  __shared__ float aS[256];
  const int tid = threadIdx.x;
  const int b   = blockIdx.x >> 6;      // 64 tiles per batch
  const int i0  = (blockIdx.x & 63) * 32;

  const float* src = Wh + ((size_t)(b * Nn + i0)) * Fn;
#pragma unroll
  for (int q = 0; q < 4; ++q) {
    const int idx = q * 1024 + tid * 4;
    const int r = idx >> 7, c = idx & 127;
    float4 v = *(const float4*)(src + idx);
    S[r * 129 + c]     = v.x;
    S[r * 129 + c + 1] = v.y;
    S[r * 129 + c + 2] = v.z;
    S[r * 129 + c + 3] = v.w;
  }
  aS[tid] = a[tid];
  __syncthreads();

  if (tid < 32) {
    float s1 = 0.f, s2 = 0.f;
    for (int c = 0; c < 128; ++c) {
      const float v = S[tid * 129 + c];
      s1 += v * aS[c];
      s2 += v * aS[128 + c];
    }
    f1[b * Nn + i0 + tid] = s1;
    f2[b * Nn + i0 + tid] = s2;
  }

  const int c  = tid >> 1;
  const int hf = tid & 1;
  __align__(16) _Float16 tmp[16];
#pragma unroll
  for (int r16 = 0; r16 < 16; ++r16)
    tmp[r16] = (_Float16)S[(hf * 16 + r16) * 129 + c];
  _Float16* dst = WhT + ((size_t)(b * Fn + c)) * Nn + i0 + hf * 16;
  ((int4*)dst)[0] = ((int4*)tmp)[0];
  ((int4*)dst)[1] = ((int4*)tmp)[1];
}

// ---------------- k3: per-batch upper bound of attention logits ------------
__global__ __launch_bounds__(256) void k3_mb(const float* __restrict__ f1,
                                             const float* __restrict__ f2,
                                             float* __restrict__ mb) {
  __shared__ float r1[256], r2[256];
  const int b = blockIdx.x, tid = threadIdx.x;
  float m1 = -3.0e38f, m2 = -3.0e38f;
  for (int idx = tid; idx < Nn; idx += 256) {
    m1 = fmaxf(m1, f1[b * Nn + idx]);
    m2 = fmaxf(m2, f2[b * Nn + idx]);
  }
  r1[tid] = m1; r2[tid] = m2;
  __syncthreads();
  for (int s = 128; s > 0; s >>= 1) {
    if (tid < s) {
      r1[tid] = fmaxf(r1[tid], r1[tid + s]);
      r2[tid] = fmaxf(r2[tid], r2[tid + s]);
    }
    __syncthreads();
  }
  if (tid == 0) {
    const float e = r1[0] + r2[0];
    mb[b] = (e > 0.f) ? e : GAT_ALPHA * e;   // LeakyReLU is monotone
  }
}

// ---------------- k4: fused masked softmax + attention @ Wh ----------------
__global__ __launch_bounds__(256) void k4_attn(const int* __restrict__ adj,
                                               const float* __restrict__ f1,
                                               const float* __restrict__ f2,
                                               const float* __restrict__ mb,
                                               const _Float16* __restrict__ WhT,
                                               float* __restrict__ out) {
  __shared__ float    f2s[Nn];           // 8 KB
  __shared__ _Float16 Btile[128 * 40];   // 10 KB, padded stride 40
  const int tid  = threadIdx.x;
  const int b    = blockIdx.x >> 4;      // 16 blocks of 128 rows per batch
  const int i0   = (blockIdx.x & 15) * 128;
  const int wave = tid >> 5;
  const int L    = tid & 31;
  const int m    = L & 15;
  const int half = L >> 4;
  const int iw   = i0 + wave * 16;
  const int row  = iw + m;

  for (int idx = tid; idx < Nn; idx += 256) f2s[idx] = f2[b * Nn + idx];

  const float f1v = f1[b * Nn + row];
  const float mbv = mb[b];
  const int* adjRow = adj + ((size_t)(b * Nn) + row) * Nn;
  const _Float16* whtB = WhT + (size_t)b * Fn * Nn;

  v8f zero8 = {0.f, 0.f, 0.f, 0.f, 0.f, 0.f, 0.f, 0.f};
  v8f acc[8];
#pragma unroll
  for (int t = 0; t < 8; ++t) acc[t] = zero8;
  float sumv = 0.f;
  __syncthreads();

  const int cst = tid >> 1, chf = tid & 1;

  for (int j0 = 0; j0 < Nn; j0 += 32) {
    // stage 128x32 f16 WhT tile into LDS (shared by all 8 waves)
    {
      const int4* s = (const int4*)(whtB + (size_t)cst * Nn + j0 + chf * 16);
      int4* d = (int4*)(&Btile[cst * 40 + chf * 16]);
      d[0] = s[0];
      d[1] = s[1];
    }
    __syncthreads();

    const int jA = j0 + half * 16;       // A layout: lane halves own K 0-15 / 16-31
    int4 a0 = *(const int4*)(adjRow + jA);
    int4 a1 = *(const int4*)(adjRow + jA + 4);
    int4 a2 = *(const int4*)(adjRow + jA + 8);
    int4 a3 = *(const int4*)(adjRow + jA + 12);
    float4 g0 = *(const float4*)(&f2s[jA]);
    float4 g1 = *(const float4*)(&f2s[jA + 4]);
    float4 g2 = *(const float4*)(&f2s[jA + 8]);
    float4 g3 = *(const float4*)(&f2s[jA + 12]);

    auto pc = [&](int ai, float f2v) -> _Float16 {
      float e = f1v + f2v;
      e = (e > 0.f) ? e : GAT_ALPHA * e;
      const float v = (ai > 0) ? e : GAT_NEGINF;
      const float p = __expf(v - mbv);   // <= 1, f16-safe
      sumv += p;
      return (_Float16)p;
    };

    v16h af;
    af[0]  = pc(a0.x, g0.x); af[1]  = pc(a0.y, g0.y);
    af[2]  = pc(a0.z, g0.z); af[3]  = pc(a0.w, g0.w);
    af[4]  = pc(a1.x, g1.x); af[5]  = pc(a1.y, g1.y);
    af[6]  = pc(a1.z, g1.z); af[7]  = pc(a1.w, g1.w);
    af[8]  = pc(a2.x, g2.x); af[9]  = pc(a2.y, g2.y);
    af[10] = pc(a2.z, g2.z); af[11] = pc(a2.w, g2.w);
    af[12] = pc(a3.x, g3.x); af[13] = pc(a3.y, g3.y);
    af[14] = pc(a3.z, g3.z); af[15] = pc(a3.w, g3.w);

    const int koff = half * 16;          // B layout halves own K 0-15 / 16-31
#pragma unroll
    for (int t = 0; t < 8; ++t) {
      v16h bf = load16h(&Btile[(t * 16 + m) * 40 + koff]);
      acc[t] = __builtin_amdgcn_wmma_f32_16x16x32_f16(
          false, af, false, bf, (short)0, acc[t], false, false);
    }
    __syncthreads();
  }

  // lanes m and m+16 (same row) together cover all j: combine partial sums
  const float rowsum = sumv + __shfl_xor(sumv, 16, 32);

#pragma unroll
  for (int r = 0; r < 8; ++r) {
    const int srcLane = r + ((L & 16) >> 1);       // C-layout row owner
    const float rs  = __shfl(rowsum, srcLane, 32);
    const float inv = 1.f / fmaxf(rs, 1e-30f);
    const int M = r + half * 8;
    float* op = out + ((size_t)(b * Nn + iw + M)) * Fn + m;
#pragma unroll
    for (int t = 0; t < 8; ++t) {
      const float x = acc[t][r] * inv;
      op[t * 16] = (x > 0.f) ? x : (__expf(x) - 1.f);   // ELU
    }
  }
}

// ---------------------------------------------------------------------------
extern "C" void kernel_launch(void* const* d_in, const int* in_sizes, int n_in,
                              void* d_out, int out_size, void* d_ws, size_t ws_size,
                              hipStream_t stream) {
  const float* h   = (const float*)d_in[0];
  const int*   adj = (const int*)d_in[1];
  const float* W   = (const float*)d_in[2];
  const float* a   = (const float*)d_in[3];
  float* out = (float*)d_out;

  char* ws = (char*)d_ws;
  float*    Wh  = (float*)(ws);                          // 8,388,608 B
  _Float16* WhT = (_Float16*)(ws + 8388608);             // 4,194,304 B
  float*    f1  = (float*)(ws + 12582912);               //    65,536 B
  float*    f2  = (float*)(ws + 12648448);               //    65,536 B
  float*    mbp = (float*)(ws + 12713984);               //        32 B

  k1_wh  <<<Bn * (Nn / 64), 128, 0, stream>>>(h, W, Wh);
  k2_prep<<<Bn * (Nn / 32), 256, 0, stream>>>(Wh, a, f1, f2, WhT);
  k3_mb  <<<Bn,             256, 0, stream>>>(f1, f2, mbp);
  k4_attn<<<Bn * (Nn / 128),256, 0, stream>>>(adj, f1, f2, mbp, WhT, out);
}
